// AdaSFMCell_74732430950689
// MI455X (gfx1250) — compile-verified, matrix operand order checked by reading, not run on hardware
//
#include <hip/hip_runtime.h>
#include <hip/hip_bf16.h>
#include <math.h>

// ---------------- types ----------------
typedef __attribute__((ext_vector_type(8)))  float          v8f;
typedef __attribute__((ext_vector_type(16))) __bf16         v16bf;
typedef __attribute__((ext_vector_type(4)))  int            v4i;
typedef __attribute__((ext_vector_type(4)))  unsigned short us4;
typedef __attribute__((ext_vector_type(8)))  unsigned short us8;
typedef __attribute__((ext_vector_type(16))) unsigned short us16;

#define U_DIM 256
#define D_DIM 256
#define B_DIM 128

// d_out layout: z (B*U) | Im (B*U*U) | Re (B*U*U) | omg (B*U)
#define Z_OFF   0
#define IM_OFF  (B_DIM * U_DIM)
#define RE_OFF  (IM_OFF + B_DIM * U_DIM * U_DIM)
#define OMG_OFF (RE_OFF + B_DIM * U_DIM * U_DIM)

// ws layout: A_s bf16 [K][B][U] (16 MB) | c (B*U f32) | ste | fre
#define WS_A_BYTES (size_t)(U_DIM * (size_t)B_DIM * U_DIM * 2)

// ---- CDNA5 async global->LDS path (guarded; falls back to b128 copy) ----
#if defined(__HIP_DEVICE_COMPILE__) && __has_builtin(__builtin_amdgcn_global_load_async_to_lds_b128)
#define HAVE_ASYNC_LDS 1
typedef __attribute__((address_space(1))) v4i g_v4i;   // global int4*
typedef __attribute__((address_space(3))) v4i l_v4i;   // LDS int4*
#else
#define HAVE_ASYNC_LDS 0
#endif

static __device__ __forceinline__ void wait_async0() {
#if defined(__HIP_DEVICE_COMPILE__)
#if __has_builtin(__builtin_amdgcn_s_wait_asynccnt)
    __builtin_amdgcn_s_wait_asynccnt(0);
#else
    asm volatile("s_wait_asynccnt 0" ::: "memory");
#endif
#endif
}

static __device__ __forceinline__ float hs(float v) {
    return fminf(fmaxf(0.2f * v + 0.5f, 0.0f), 1.0f);
}

static __device__ __forceinline__ unsigned short f2bf(float f) {
    unsigned int u = __float_as_uint(f);
    u = (u + 0x7FFFu + ((u >> 16) & 1u)) >> 16;   // round-to-nearest-even
    return (unsigned short)u;
}

// Build a v16bf WMMA operand from an LDS row: K = kh..kh+7 then kh+16..kh+23
static __device__ __forceinline__ v16bf ld_op(const unsigned short* rowbase, int kh) {
    us8 lo = *(const us8*)(rowbase + kh);
    us8 hi = *(const us8*)(rowbase + kh + 16);
    us16 c;
#pragma unroll
    for (int i = 0; i < 8; ++i) { c[i] = lo[i]; c[i + 8] = hi[i]; }
    return __builtin_bit_cast(v16bf, c);
}

// ---------------- kernel 0: zero z region ----------------
__global__ void k0_zero_z(float* __restrict__ out) {
    int idx = blockIdx.x * 256 + threadIdx.x;          // 128 * 256 = 32768
    out[Z_OFF + idx] = 0.0f;
}

// ---------------- kernel 1: gates ----------------
__global__ void k1_gates(const float* __restrict__ x, const float* __restrict__ z_prev,
                         const float* __restrict__ Wk, const float* __restrict__ Wr,
                         const float* __restrict__ bias,
                         float* __restrict__ out,
                         float* __restrict__ c_ws, float* __restrict__ ste_ws,
                         float* __restrict__ fre_ws) {
    __shared__ float xs[D_DIM];
    __shared__ float zs[U_DIM];
    const int b = blockIdx.x;
    const int col = threadIdx.x;
    xs[col] = x[b * D_DIM + col];
    zs[col] = z_prev[b * U_DIM + col];
    __syncthreads();

    float a0 = bias[col];
    float a1 = bias[U_DIM + col];
    float a2 = bias[2 * U_DIM + col];
    float a3 = bias[3 * U_DIM + col];
    float a4 = bias[4 * U_DIM + col];

#pragma unroll 4
    for (int d = 0; d < D_DIM; ++d) {
        const float xv = xs[d];
        const float* kr = Wk + (size_t)d * (5 * U_DIM) + col;
        a0 += xv * kr[0];
        a1 += xv * kr[U_DIM];
        a2 += xv * kr[2 * U_DIM];
        a3 += xv * kr[3 * U_DIM];
        a4 += xv * kr[4 * U_DIM];
    }
#pragma unroll 4
    for (int u = 0; u < U_DIM; ++u) {
        const float zv = zs[u];
        const float* rr = Wr + (size_t)u * (5 * U_DIM) + col;
        a0 += zv * rr[0];
        a1 += zv * rr[U_DIM];
        a2 += zv * rr[2 * U_DIM];
        a3 += zv * rr[3 * U_DIM];
        a4 += zv * rr[4 * U_DIM];
    }
    const float ig  = hs(a0);
    const float fre = hs(a1);
    const float ste = hs(a2);
    const float g   = tanhf(a3);
    const int o = b * U_DIM + col;
    c_ws[o]   = ig * g;
    ste_ws[o] = ste;
    fre_ws[o] = fre;
    out[OMG_OFF + o] = a4;       // omg = x_omg + h_omg
}

// ---------------- kernel 2: Re / Im / A ----------------
__global__ void k2_recur(const float* __restrict__ Re_prev, const float* __restrict__ Im_prev,
                         const float* __restrict__ omg_prev, const float* __restrict__ t_in,
                         const float* __restrict__ c_ws, const float* __restrict__ ste_ws,
                         const float* __restrict__ fre_ws,
                         float* __restrict__ out, unsigned short* __restrict__ A_s) {
    __shared__ float sCos[64], sSin[64], sFre[64], sSte[64], sC[64];
    __shared__ __align__(16) unsigned short tA[64][72];

    const int b  = blockIdx.x;
    const int u0 = blockIdx.y * 64;
    const int k0 = blockIdx.z * 64;
    const int tid = threadIdx.x;

    if (tid < 64) {
        const int k = k0 + tid;
        const float th = omg_prev[b * U_DIM + k] * t_in[b];
        sCos[tid] = cosf(th);
        sSin[tid] = sinf(th);
        sFre[tid] = fre_ws[b * U_DIM + k];
    } else if (tid < 128) {
        const int u = u0 + (tid - 64);
        sSte[tid - 64] = ste_ws[b * U_DIM + u];
        sC[tid - 64]   = c_ws[b * U_DIM + u];
    }
    __syncthreads();

#pragma unroll
    for (int i = 0; i < 16; ++i) {
        const int idx = tid + i * 256;           // 4096 elements
        const int up = idx >> 6;
        const int kp = idx & 63;
        const size_t g = (size_t)b * U_DIM * U_DIM + (size_t)(u0 + up) * U_DIM + (k0 + kp);
        const float f = sSte[up] * sFre[kp];
        const float re = f * Re_prev[g] + sC[up] * sCos[kp];
        const float im = f * Im_prev[g] + sC[up] * sSin[kp];
        out[RE_OFF + g] = re;
        out[IM_OFF + g] = im;
        tA[kp][up] = f2bf(sqrtf(re * re + im * im));
    }
    __syncthreads();

#pragma unroll
    for (int i = 0; i < 16; ++i) {
        const int idx = tid + i * 256;
        const int kp = idx >> 6;
        const int up = idx & 63;
        A_s[(size_t)(k0 + kp) * (B_DIM * U_DIM) + (size_t)b * U_DIM + (u0 + up)] = tA[kp][up];
    }
}

// ---------------- kernel 3: batched frequency GEMMs (WMMA) ----------------
__global__ void __launch_bounds__(256)
k3_freq(const unsigned short* __restrict__ A_s, const float* __restrict__ x,
        const float* __restrict__ z_prev, const float* __restrict__ freq_k,
        const float* __restrict__ freq_k_input, const float* __restrict__ freq_bias,
        float* __restrict__ out) {
    __shared__ __align__(16) unsigned short lA[B_DIM][40];
    __shared__ __align__(16) unsigned short lX[B_DIM][40];
    __shared__ __align__(16) unsigned short lZ[B_DIM][40];
    __shared__ __align__(16) unsigned short lBu[64][40];   // U_o^T tile
    __shared__ __align__(16) unsigned short lBw[64][40];   // W_z^T tile
    __shared__ __align__(16) unsigned short lBo[64][40];   // W_o^T tile
    __shared__ __align__(16) unsigned short lBv[64][40];   // V_o^T tile

    const int k   = blockIdx.x;
    const int v0  = blockIdx.y * 64;
    const int tid = threadIdx.x;
    const int lane = tid & 31;
    const int wv   = tid >> 5;
    const int m0   = wv * 16;
    const int row  = lane & 15;
    const int kh   = (lane < 16) ? 0 : 8;

    v8f oacc[4] = {{}, {}, {}, {}};
    v8f zacc[4] = {{}, {}, {}, {}};

    for (int kc = 0; kc < U_DIM / 32; ++kc) {
        const int ub = kc * 32;

        // --- stage A_s chunk [128][32] bf16: pure byte copy -> async to LDS ---
#pragma unroll
        for (int i = 0; i < 2; ++i) {
            const int cid = tid + i * 256;                 // 512 x 16B chunks
            const int bb = cid >> 2;
            const int q  = (cid & 3) * 8;
            const unsigned short* gp =
                A_s + (size_t)k * (B_DIM * U_DIM) + (size_t)bb * U_DIM + ub + q;
            unsigned short* lp = &lA[bb][q];
#if HAVE_ASYNC_LDS
            __builtin_amdgcn_global_load_async_to_lds_b128(
                (g_v4i*)(void*)gp, (l_v4i*)(void*)lp, 0, 0);
#else
            *(us8*)lp = *(const us8*)gp;
#endif
        }

        // --- stage x, z chunks: b128 reads, packed 8B LDS stores ---
#pragma unroll
        for (int i = 0; i < 4; ++i) {
            const int cid = tid + i * 256;                 // 1024 x 4-elem chunks
            const int bb = cid >> 3;
            const int uq = (cid & 7) * 4;
            const float4 xv = *(const float4*)&x[(size_t)bb * D_DIM + ub + uq];
            const float4 zv = *(const float4*)&z_prev[(size_t)bb * U_DIM + ub + uq];
            us4 xp, zp;
            xp[0] = f2bf(xv.x); xp[1] = f2bf(xv.y); xp[2] = f2bf(xv.z); xp[3] = f2bf(xv.w);
            zp[0] = f2bf(zv.x); zp[1] = f2bf(zv.y); zp[2] = f2bf(zv.z); zp[3] = f2bf(zv.w);
            *(us4*)&lX[bb][uq] = xp;
            *(us4*)&lZ[bb][uq] = zp;
        }

        // --- stage transposed weight tiles [64 v][32 u]: b128 reads ---
#pragma unroll
        for (int i = 0; i < 2; ++i) {
            const int cid = tid + i * 256;                 // 512 x 4-elem chunks
            const int uu = cid >> 4;
            const int vq = (cid & 15) * 4;
            const size_t fk = ((size_t)k * U_DIM + (ub + uu)) * (3 * U_DIM) + v0 + vq;
            const float4 wu = *(const float4*)&freq_k[fk];                 // U_o
            const float4 wv4 = *(const float4*)&freq_k[fk + U_DIM];        // V_o
            const float4 ww = *(const float4*)&freq_k[fk + 2 * U_DIM];     // W_z
            const float4 wo = *(const float4*)
                &freq_k_input[((size_t)k * D_DIM + (ub + uu)) * U_DIM + v0 + vq];
            lBu[vq + 0][uu] = f2bf(wu.x); lBu[vq + 1][uu] = f2bf(wu.y);
            lBu[vq + 2][uu] = f2bf(wu.z); lBu[vq + 3][uu] = f2bf(wu.w);
            lBv[vq + 0][uu] = f2bf(wv4.x); lBv[vq + 1][uu] = f2bf(wv4.y);
            lBv[vq + 2][uu] = f2bf(wv4.z); lBv[vq + 3][uu] = f2bf(wv4.w);
            lBw[vq + 0][uu] = f2bf(ww.x); lBw[vq + 1][uu] = f2bf(ww.y);
            lBw[vq + 2][uu] = f2bf(ww.z); lBw[vq + 3][uu] = f2bf(ww.w);
            lBo[vq + 0][uu] = f2bf(wo.x); lBo[vq + 1][uu] = f2bf(wo.y);
            lBo[vq + 2][uu] = f2bf(wo.z); lBo[vq + 3][uu] = f2bf(wo.w);
        }

        // --- prefetch next K-chunk's weight rows (overlaps WMMA phase) ---
        if (kc + 1 < U_DIM / 32) {
            const int ubn = ub + 32;
            const int uu = tid >> 3;                       // 0..31
            const int vv = (tid & 7) * 8;
            const size_t fkn = ((size_t)k * U_DIM + (ubn + uu)) * (3 * U_DIM) + v0 + vv;
            __builtin_prefetch(&freq_k[fkn], 0, 0);
            __builtin_prefetch(&freq_k[fkn + 2 * U_DIM], 0, 0);
            __builtin_prefetch(&freq_k_input[((size_t)k * D_DIM + (ubn + uu)) * U_DIM + v0 + vv], 0, 0);
        }

#if HAVE_ASYNC_LDS
        wait_async0();
#endif
        __syncthreads();

        const v16bf va = ld_op(&lA[m0 + row][0], kh);
        const v16bf vx = ld_op(&lX[m0 + row][0], kh);
        const v16bf vz = ld_op(&lZ[m0 + row][0], kh);
#pragma unroll
        for (int j = 0; j < 4; ++j) {
            const int nl = j * 16 + row;
            const v16bf bu = ld_op(&lBu[nl][0], kh);
            const v16bf bo = ld_op(&lBo[nl][0], kh);
            const v16bf bv = ld_op(&lBv[nl][0], kh);
            const v16bf bw = ld_op(&lBw[nl][0], kh);
            oacc[j] = __builtin_amdgcn_wmma_f32_16x16x32_bf16(false, va, false, bu,
                                                              (short)0, oacc[j], false, false);
            oacc[j] = __builtin_amdgcn_wmma_f32_16x16x32_bf16(false, vx, false, bo,
                                                              (short)0, oacc[j], false, false);
            oacc[j] = __builtin_amdgcn_wmma_f32_16x16x32_bf16(false, vz, false, bv,
                                                              (short)0, oacc[j], false, false);
            zacc[j] = __builtin_amdgcn_wmma_f32_16x16x32_bf16(false, va, false, bw,
                                                              (short)0, zacc[j], false, false);
        }
        __syncthreads();
    }

    // --- epilogue: biases, activations, reduce over k into z via atomics ---
    const int hi8 = (lane & 16) ? 8 : 0;
#pragma unroll
    for (int j = 0; j < 4; ++j) {
        const int v = v0 + j * 16 + row;
        const float bo = freq_bias[(size_t)k * U_DIM + v];               // b_o[k][v]
        const float bz = freq_bias[(size_t)(U_DIM + k) * U_DIM + v];     // b_z[k][v]
#pragma unroll
        for (int r = 0; r < 8; ++r) {
            const int b = m0 + hi8 + r;
            const float o  = hs(oacc[j][r] + bo);
            const float zz = o * tanhf(zacc[j][r] + bz);
            atomicAdd(out + Z_OFF + (size_t)b * U_DIM + v, zz);
        }
    }
}

// ---------------- launcher ----------------
extern "C" void kernel_launch(void* const* d_in, const int* in_sizes, int n_in,
                              void* d_out, int out_size, void* d_ws, size_t ws_size,
                              hipStream_t stream) {
    const float* x        = (const float*)d_in[0];
    const float* t        = (const float*)d_in[1];
    const float* z_prev   = (const float*)d_in[2];
    const float* Im_prev  = (const float*)d_in[3];
    const float* Re_prev  = (const float*)d_in[4];
    const float* omg_prev = (const float*)d_in[5];
    const float* Wk       = (const float*)d_in[6];
    const float* Wr       = (const float*)d_in[7];
    const float* freq_k   = (const float*)d_in[8];
    const float* freq_ki  = (const float*)d_in[9];
    const float* bias     = (const float*)d_in[10];
    const float* fbias    = (const float*)d_in[11];

    float* out = (float*)d_out;
    unsigned short* A_s = (unsigned short*)d_ws;
    float* c_ws   = (float*)((char*)d_ws + WS_A_BYTES);
    float* ste_ws = c_ws + B_DIM * U_DIM;
    float* fre_ws = ste_ws + B_DIM * U_DIM;

    k0_zero_z<<<dim3(B_DIM), dim3(256), 0, stream>>>(out);
    k1_gates<<<dim3(B_DIM), dim3(256), 0, stream>>>(x, z_prev, Wk, Wr, bias,
                                                    out, c_ws, ste_ws, fre_ws);
    k2_recur<<<dim3(B_DIM, U_DIM / 64, U_DIM / 64), dim3(256), 0, stream>>>(
        Re_prev, Im_prev, omg_prev, t, c_ws, ste_ws, fre_ws, out, A_s);
    k3_freq<<<dim3(U_DIM, U_DIM / 64), dim3(256), 0, stream>>>(
        A_s, x, z_prev, freq_k, freq_ki, fbias, out);
}
